// MLABlock_19155554140860
// MI455X (gfx1250) — matrix operand order repaired and checked by reading, not compiled, not run on hardware
//
#include <hip/hip_runtime.h>
#include <hip/hip_bf16.h>

// ---------------- problem constants ----------------
constexpr int Bc = 2, Sc = 1024, Dc = 2048;
constexpr int Hc = 16, DNc = 128, DRc = 64, DVc = 128, Rc = 512;
constexpr int Ec = 12, Kt = 4, FIc = 1408, SIc = 2816;
constexpr int Tc  = Bc * Sc;            // 2048 tokens
constexpr int BH  = Bc * Hc;            // 32 (batch*heads)
constexpr int QD  = DNc + DRc;          // 192
constexpr int QW  = Hc * QD;            // 3072
constexpr int KVA = Rc + DRc;           // 576  (logical width of kv row)
constexpr int KVAP = 640;               // padded kv row stride (multiple of 128)
constexpr int GEP  = 128;               // padded gate width
constexpr int KVB = Hc * (DNc + DVc);   // 4096

typedef __attribute__((ext_vector_type(16))) _Float16 v16h;
typedef __attribute__((ext_vector_type(2)))  __fp16   v2fp16;
typedef __attribute__((ext_vector_type(8)))  float    v8f;
typedef __attribute__((ext_vector_type(4)))  float    f32x4;

union Frag {
  v16h h;
  unsigned int u[8];
};

// ---------------- generic batched WMMA GEMM ----------------
// PRECONDITIONS (guaranteed by the launcher): M % 128 == 0, N % 128 == 0,
// K % 32 == 0. No bounds checks anywhere -> pure vector loads + WMMA.
// Block tile 128x128, TK=32, double-buffered LDS, 8 waves each computing a
// 32x64 strip: 8 x v_wmma_f32_16x16x32_f16 per wave per k-chunk.
#define TM 128
#define TN 128
#define TK 32

// one v_cvt_pk_rtz_f16_f32 instead of 2x cvt + shift + or
__device__ __forceinline__ unsigned int pack2(float a, float b) {
  union { v2fp16 h; unsigned int u; } t;
  t.h = __builtin_amdgcn_cvt_pkrtz(a, b);
  return t.u;
}

__device__ __forceinline__ void stage_tiles(
    const float* __restrict__ A, const float* __restrict__ B,
    int lda, int ldb, int m0, int n0, int k0, int tid,
    _Float16 (* __restrict__ As)[TK], _Float16 (* __restrict__ Bs)[TK])
{
  // ---- A tile: 128 rows x 32 k, row-major f16 ----
  {
    const int row = tid >> 1;
    const int kb  = (tid & 1) * 16;
    const f32x4* src = (const f32x4*)(A + (long long)(m0 + row) * lda + (k0 + kb));
    const f32x4 f0 = src[0], f1 = src[1], f2 = src[2], f3 = src[3];
    unsigned int* dst = (unsigned int*)&As[row][kb];
    dst[0] = pack2(f0.x, f0.y); dst[1] = pack2(f0.z, f0.w);
    dst[2] = pack2(f1.x, f1.y); dst[3] = pack2(f1.z, f1.w);
    dst[4] = pack2(f2.x, f2.y); dst[5] = pack2(f2.z, f2.w);
    dst[6] = pack2(f3.x, f3.y); dst[7] = pack2(f3.z, f3.w);
  }
  // ---- B tile: 32 k x 128 n, stored transposed [n][k], K-pairs packed ----
#pragma unroll
  for (int it = 0; it < 2; ++it) {
    const int item = tid + it * 256;         // 0..511
    const int kp   = item >> 5;              // 0..15  (k-pair index)
    const int nb   = (item & 31) * 4;        // 0..124
    const int gk   = k0 + 2 * kp;
    const f32x4 r0 = *(const f32x4*)(B + (long long)gk * ldb + (n0 + nb));
    const f32x4 r1 = *(const f32x4*)(B + (long long)(gk + 1) * ldb + (n0 + nb));
    *(unsigned int*)&Bs[nb + 0][2 * kp] = pack2(r0.x, r1.x);
    *(unsigned int*)&Bs[nb + 1][2 * kp] = pack2(r0.y, r1.y);
    *(unsigned int*)&Bs[nb + 2][2 * kp] = pack2(r0.z, r1.z);
    *(unsigned int*)&Bs[nb + 3][2 * kp] = pack2(r0.w, r1.w);
  }
}

__global__ __launch_bounds__(256) void gemm_wmma(
    const float* __restrict__ A, const float* __restrict__ B,
    float* __restrict__ C, int K,
    int lda, int ldb, int ldc,
    long long strideA, long long strideB, long long strideC)
{
  __shared__ _Float16 As[2][TM][TK];
  __shared__ _Float16 Bs[2][TN][TK];

  const int z = blockIdx.z;
  A += (long long)z * strideA;
  B += (long long)z * strideB;
  C += (long long)z * strideC;

  const int m0   = blockIdx.y * TM;
  const int n0   = blockIdx.x * TN;
  const int tid  = threadIdx.x;
  const int lane = tid & 31;
  const int wave = tid >> 5;
  const int wr   = (wave >> 1) * 32;   // 4 wave-rows x 32
  const int wc   = (wave & 1) * 64;    // 2 wave-cols x 64
  const int half = lane >> 4;
  const int r    = lane & 15;

  v8f acc[2][4];
#pragma unroll
  for (int i = 0; i < 2; ++i)
#pragma unroll
    for (int j = 0; j < 4; ++j)
      acc[i][j] = (v8f){};

  const int nk = K / TK;
  stage_tiles(A, B, lda, ldb, m0, n0, 0, tid, As[0], Bs[0]);
  __syncthreads();

  for (int kk = 0; kk < nk; ++kk) {
    const int cur = kk & 1;
    if (kk + 1 < nk)
      stage_tiles(A, B, lda, ldb, m0, n0, (kk + 1) * TK, tid,
                  As[cur ^ 1], Bs[cur ^ 1]);
    if (kk + 2 < nk) {
      __builtin_prefetch(A + (long long)(m0 + (tid >> 1)) * lda + (kk + 2) * TK, 0, 3);
      __builtin_prefetch(B + (long long)((kk + 2) * TK + (tid >> 3)) * ldb + n0, 0, 3);
    }

    // ---- fragments from current buffer ----
    Frag fa[2], fb[4];
#pragma unroll
    for (int i = 0; i < 2; ++i) {
      const unsigned int* ArowU = (const unsigned int*)&As[cur][wr + i * 16 + r][0];
#pragma unroll
      for (int vv = 0; vv < 8; ++vv) {
        const int dw = vv + half * 8;
        fa[i].u[vv] = ArowU[dw];
      }
    }
#pragma unroll
    for (int j = 0; j < 4; ++j) {
      const unsigned int* BcolU = (const unsigned int*)&Bs[cur][wc + j * 16 + r][0];
#pragma unroll
      for (int vv = 0; vv < 8; ++vv) fb[j].u[vv] = BcolU[half * 8 + vv];
    }

#pragma unroll
    for (int i = 0; i < 2; ++i)
#pragma unroll
      for (int j = 0; j < 4; ++j)
        acc[i][j] = __builtin_amdgcn_wmma_f32_16x16x32_f16(
            false, fa[i].h, false, fb[j].h, (short)0, acc[i][j], false, false);

    __syncthreads();
  }

  // ---- store C: within a 16x16 tile, VGPR vv -> M=vv+8*half, N=lane&15 ----
#pragma unroll
  for (int i = 0; i < 2; ++i) {
#pragma unroll
    for (int j = 0; j < 4; ++j) {
      const int n = n0 + wc + j * 16 + r;
#pragma unroll
      for (int vv = 0; vv < 8; ++vv) {
        const int m = m0 + wr + i * 16 + vv + 8 * half;
        C[(long long)m * ldc + n] = acc[i][j][vv];
      }
    }
  }
}

// ---------------- elementwise / reduction kernels ----------------
// zero-pad columns: dst[r][0..dst_cols) = (c < src_cols) ? src[r][c] : 0
__global__ void pad_kernel(const float* __restrict__ src, float* __restrict__ dst,
                           int src_cols, int dst_cols, long long n)
{
  long long i = (long long)blockIdx.x * blockDim.x + threadIdx.x;
  if (i >= n) return;
  const int c = i % dst_cols;
  const long long rr = i / dst_cols;
  dst[i] = (c < src_cols) ? src[rr * src_cols + c] : 0.0f;
}

__global__ __launch_bounds__(256) void rmsnorm_kernel(
    const float* __restrict__ x, const float* __restrict__ w,
    float* __restrict__ out, int cols, int in_stride, int out_stride)
{
  const int row = blockIdx.x;
  const float* xr = x + (long long)row * in_stride;
  float* orow = out + (long long)row * out_stride;
  __shared__ float red[256];
  float s = 0.f;
  for (int c = threadIdx.x; c < cols; c += 256) { float v = xr[c]; s += v * v; }
  red[threadIdx.x] = s;
  __syncthreads();
  for (int off = 128; off > 0; off >>= 1) {
    if (threadIdx.x < off) red[threadIdx.x] += red[threadIdx.x + off];
    __syncthreads();
  }
  const float inv = rsqrtf(red[0] / (float)cols + 1e-6f);
  for (int c = threadIdx.x; c < cols; c += 256) orow[c] = xr[c] * inv * w[c];
}

__global__ void rope_q_kernel(float* __restrict__ q, const int* __restrict__ sp)
{
  int idx = blockIdx.x * blockDim.x + threadIdx.x;
  const int NP = DRc / 2;
  if (idx >= Tc * Hc * NP) return;
  const int i = idx % NP;
  const int h = (idx / NP) % Hc;
  const int t = idx / (NP * Hc);
  const int s = t % Sc;
  const float pos = (float)(*sp + s);
  const float inv = __powf(10000.0f, -(float)(2 * i) / (float)DRc);
  float sn, cs;
  __sincosf(pos * inv, &sn, &cs);
  const long long base = (long long)t * QW + h * QD + DNc;
  const float x1 = q[base + 2 * i], x2 = q[base + 2 * i + 1];
  q[base + 2 * i]     = x1 * cs - x2 * sn;
  q[base + 2 * i + 1] = x1 * sn + x2 * cs;
}

// kv rows have stride KVAP (padded); rope region at [Rc, Rc+DRc)
__global__ void rope_k_kernel(float* __restrict__ kv, const int* __restrict__ sp)
{
  int idx = blockIdx.x * blockDim.x + threadIdx.x;
  const int NP = DRc / 2;
  if (idx >= Tc * NP) return;
  const int i = idx % NP;
  const int t = idx / NP;
  const int s = t % Sc;
  const float pos = (float)(*sp + s);
  const float inv = __powf(10000.0f, -(float)(2 * i) / (float)DRc);
  float sn, cs;
  __sincosf(pos * inv, &sn, &cs);
  const long long base = (long long)t * KVAP + Rc;
  const float x1 = kv[base + 2 * i], x2 = kv[base + 2 * i + 1];
  kv[base + 2 * i]     = x1 * cs - x2 * sn;
  kv[base + 2 * i + 1] = x1 * sn + x2 * cs;
}

// qpack[bh][s][d] (d<192), from q[t][h*192+d] (rope already applied)
__global__ void pack_q_kernel(const float* __restrict__ q, float* __restrict__ qp)
{
  long long idx = (long long)blockIdx.x * blockDim.x + threadIdx.x;
  if (idx >= (long long)BH * Sc * QD) return;
  const int d  = idx % QD;
  const int s  = (idx / QD) % Sc;
  const int bh = idx / ((long long)QD * Sc);
  const int b  = bh / Hc, h = bh % Hc;
  qp[idx] = q[((long long)(b * Sc + s)) * QW + h * QD + d];
}

// kpack[bh][d][t] (K-major, i.e. transposed): d<128 from kvb k_nope, else k_pe
__global__ void pack_k_kernel(const float* __restrict__ kvb,
                              const float* __restrict__ kv,
                              float* __restrict__ kp)
{
  long long idx = (long long)blockIdx.x * blockDim.x + threadIdx.x;
  if (idx >= (long long)BH * QD * Sc) return;
  const int t  = idx % Sc;
  const int d  = (idx / Sc) % QD;
  const int bh = idx / ((long long)Sc * QD);
  const int b  = bh / Hc, h = bh % Hc;
  float v;
  if (d < DNc)
    v = kvb[((long long)(b * Sc + t)) * KVB + h * (DNc + DVc) + d];
  else
    v = kv[((long long)(b * Sc + t)) * KVAP + Rc + (d - DNc)];
  kp[idx] = v;
}

// vpack[bh][t][d]
__global__ void pack_v_kernel(const float* __restrict__ kvb, float* __restrict__ vp)
{
  long long idx = (long long)blockIdx.x * blockDim.x + threadIdx.x;
  if (idx >= (long long)BH * Sc * DVc) return;
  const int d  = idx % DVc;
  const int t  = (idx / DVc) % Sc;
  const int bh = idx / ((long long)DVc * Sc);
  const int b  = bh / Hc, h = bh % Hc;
  vp[idx] = kvb[((long long)(b * Sc + t)) * KVB + h * (DNc + DVc) + DNc + d];
}

// o[t][h*128+d] <- ohead[bh][s][d]
__global__ void unpack_o_kernel(const float* __restrict__ oh, float* __restrict__ o)
{
  long long idx = (long long)blockIdx.x * blockDim.x + threadIdx.x;
  if (idx >= (long long)BH * Sc * DVc) return;
  const int d  = idx % DVc;
  const int s  = (idx / DVc) % Sc;
  const int bh = idx / ((long long)DVc * Sc);
  const int b  = bh / Hc, h = bh % Hc;
  o[((long long)(b * Sc + s)) * (Hc * DVc) + h * DVc + d] = oh[idx];
}

// causal masked softmax over rows of scores[bh][s][:]
__global__ __launch_bounds__(256) void softmax_kernel(float* __restrict__ scores,
                                                      float scale)
{
  const int row = blockIdx.x;          // bh*S + s
  const int s   = row % Sc;
  float* p = scores + (long long)row * Sc;
  __shared__ float red[256];
  float mx = -1e30f;
  for (int t = threadIdx.x; t < Sc; t += 256) {
    float v = p[t] * scale + (t > s ? -1e9f : 0.f);
    p[t] = v;
    mx = fmaxf(mx, v);
  }
  red[threadIdx.x] = mx;
  __syncthreads();
  for (int off = 128; off > 0; off >>= 1) {
    if (threadIdx.x < off) red[threadIdx.x] = fmaxf(red[threadIdx.x], red[threadIdx.x + off]);
    __syncthreads();
  }
  mx = red[0];
  __syncthreads();
  float sum = 0.f;
  for (int t = threadIdx.x; t < Sc; t += 256) {
    float e = __expf(p[t] - mx);
    p[t] = e;
    sum += e;
  }
  red[threadIdx.x] = sum;
  __syncthreads();
  for (int off = 128; off > 0; off >>= 1) {
    if (threadIdx.x < off) red[threadIdx.x] += red[threadIdx.x + off];
    __syncthreads();
  }
  const float inv = 1.0f / red[0];
  for (int t = threadIdx.x; t < Sc; t += 256) p[t] *= inv;
}

__global__ void add2_kernel(const float* a, const float* b, float* o, long long n)
{
  long long i = (long long)blockIdx.x * blockDim.x + threadIdx.x;
  if (i < n) o[i] = a[i] + b[i];
}

__global__ void add3_kernel(const float* a, const float* b, const float* c,
                            float* o, long long n)
{
  long long i = (long long)blockIdx.x * blockDim.x + threadIdx.x;
  if (i < n) o[i] = a[i] + b[i] + c[i];
}

__global__ void fill0_kernel(float* p, long long n)
{
  long long i = (long long)blockIdx.x * blockDim.x + threadIdx.x;
  if (i < n) p[i] = 0.f;
}

__global__ void silu_mul_kernel(const float* __restrict__ a,
                                const float* __restrict__ b,
                                float* __restrict__ o, long long n)
{
  long long i = (long long)blockIdx.x * blockDim.x + threadIdx.x;
  if (i < n) {
    float x = a[i];
    o[i] = (x / (1.0f + __expf(-x))) * b[i];
  }
}

// sigmoid gating + top-4 select (bias added for selection only) + normalize
// gates rows have stride GEP (padded)
__global__ void route_kernel(const float* __restrict__ gates,
                             const float* __restrict__ gate_b,
                             float* __restrict__ route)
{
  int t = blockIdx.x * blockDim.x + threadIdx.x;
  if (t >= Tc) return;
  float sg[Ec], biased[Ec], wgt[Ec];
  bool sel[Ec];
  for (int e = 0; e < Ec; ++e) {
    float v = 1.0f / (1.0f + __expf(-gates[(long long)t * GEP + e]));
    sg[e] = v;
    biased[e] = v + gate_b[e];
    sel[e] = false;
    wgt[e] = 0.f;
  }
  float wsum = 0.f;
  for (int k = 0; k < Kt; ++k) {
    int best = 0;
    float bv = -1e30f;
    for (int e = 0; e < Ec; ++e)
      if (!sel[e] && biased[e] > bv) { bv = biased[e]; best = e; }
    sel[best] = true;
    wgt[best] = sg[best];
    wsum += sg[best];
  }
  const float inv = 1.0f / wsum;
  for (int e = 0; e < Ec; ++e) route[t * Ec + e] = wgt[e] * inv;
}

// moe_out[t][d] += route[t][e] * eo[t][d]
__global__ void route_accum_kernel(const float* __restrict__ eo,
                                   const float* __restrict__ route,
                                   float* __restrict__ out, int e)
{
  long long i = (long long)blockIdx.x * blockDim.x + threadIdx.x;
  if (i >= (long long)Tc * Dc) return;
  const int t = i / Dc;
  const float w = route[t * Ec + e];
  out[i] += w * eo[i];
}

// ---------------- launcher ----------------
static inline int cdiv(int a, int b) { return (a + b - 1) / b; }

static void gemm(const float* A, const float* B, float* C,
                 int M, int N, int K, int lda, int ldb, int ldc,
                 long long sA, long long sB, long long sC, int batch,
                 hipStream_t stream)
{
  // M % 128 == 0, N % 128 == 0, K % 32 == 0 guaranteed by callers
  dim3 grid(N / TN, M / TM, batch);
  gemm_wmma<<<grid, 256, 0, stream>>>(A, B, C, K, lda, ldb, ldc, sA, sB, sC);
}

extern "C" void kernel_launch(void* const* d_in, const int* in_sizes, int n_in,
                              void* d_out, int out_size, void* d_ws, size_t ws_size,
                              hipStream_t stream)
{
  (void)in_sizes; (void)n_in; (void)out_size; (void)ws_size;
  const float* x       = (const float*)d_in[0];
  const float* wq      = (const float*)d_in[1];
  const float* wkv_a   = (const float*)d_in[2];
  const float* kvn_w   = (const float*)d_in[3];
  const float* wkv_b   = (const float*)d_in[4];
  const float* wo      = (const float*)d_in[5];
  const float* norm_w  = (const float*)d_in[6];
  const float* gate_w  = (const float*)d_in[7];
  const float* gate_b  = (const float*)d_in[8];
  const float* w1      = (const float*)d_in[9];
  const float* w2      = (const float*)d_in[10];
  const float* w3      = (const float*)d_in[11];
  const float* ws1     = (const float*)d_in[12];
  const float* ws2     = (const float*)d_in[13];
  const float* ws3     = (const float*)d_in[14];
  /* d_in[15] = mask (causal handled analytically) */
  const int*   sp      = (const int*)d_in[16];
  float* out = (float*)d_out;

  // carve workspace
  char* p = (char*)d_ws;
  auto carve = [&](long long nf) { float* r = (float*)p; p += nf * sizeof(float); return r; };
  float* h1     = carve((long long)Tc * Dc);
  float* q      = carve((long long)Tc * QW);
  float* kv     = carve((long long)Tc * KVAP);   // padded stride 640
  float* kvlat  = carve((long long)Tc * Rc);
  float* kvb    = carve((long long)Tc * KVB);
  float* qpack  = carve((long long)BH * Sc * QD);
  float* kpack  = carve((long long)BH * QD * Sc);
  float* vpack  = carve((long long)BH * Sc * DVc);
  float* scores = carve((long long)BH * Sc * Sc);
  float* ohead  = carve((long long)BH * Sc * DVc);
  float* o      = carve((long long)Tc * Dc);
  float* first  = carve((long long)Tc * Dc);
  float* h2     = carve((long long)Tc * Dc);
  float* gates  = carve((long long)Tc * GEP);    // padded stride 128
  float* route  = carve((long long)Tc * Ec);
  float* hh     = carve((long long)Tc * FIc);
  float* gg3    = carve((long long)Tc * FIc);
  float* eo     = carve((long long)Tc * Dc);
  float* moeo   = carve((long long)Tc * Dc);
  float* sa     = carve((long long)Tc * SIc);
  float* sb     = carve((long long)Tc * SIc);
  float* sho    = carve((long long)Tc * Dc);
  float* wkvap  = carve((long long)Dc * KVAP);   // zero-padded wkv_a
  float* gwpad  = carve((long long)Dc * GEP);    // zero-padded gate_w

  const long long TD = (long long)Tc * Dc;

  // ---- pad irregular-N weights so every GEMM is exactly tile-aligned ----
  {
    const long long nA = (long long)Dc * KVAP;
    pad_kernel<<<cdiv((int)nA, 256), 256, 0, stream>>>(wkv_a, wkvap, KVA, KVAP, nA);
    const long long nG = (long long)Dc * GEP;
    pad_kernel<<<cdiv((int)nG, 256), 256, 0, stream>>>(gate_w, gwpad, Ec, GEP, nG);
  }

  // ---- MLA ----
  rmsnorm_kernel<<<Tc, 256, 0, stream>>>(x, norm_w, h1, Dc, Dc, Dc);
  gemm(h1, wq,    q,  Tc, QW,   Dc, Dc, QW,   QW,   0, 0, 0, 1, stream);
  gemm(h1, wkvap, kv, Tc, KVAP, Dc, Dc, KVAP, KVAP, 0, 0, 0, 1, stream);
  rmsnorm_kernel<<<Tc, 256, 0, stream>>>(kv, kvn_w, kvlat, Rc, KVAP, Rc);
  gemm(kvlat, wkv_b, kvb, Tc, KVB, Rc, Rc, KVB, KVB, 0, 0, 0, 1, stream);

  rope_q_kernel<<<cdiv(Tc * Hc * (DRc / 2), 256), 256, 0, stream>>>(q, sp);
  rope_k_kernel<<<cdiv(Tc * (DRc / 2), 256), 256, 0, stream>>>(kv, sp);

  pack_q_kernel<<<cdiv(BH * Sc * QD, 256), 256, 0, stream>>>(q, qpack);
  pack_k_kernel<<<cdiv(BH * QD * Sc, 256), 256, 0, stream>>>(kvb, kv, kpack);
  pack_v_kernel<<<cdiv(BH * Sc * DVc, 256), 256, 0, stream>>>(kvb, vpack);

  // scores[bh] = qpack[bh] (S x 192) @ kpack[bh] (192 x S)
  gemm(qpack, kpack, scores, Sc, Sc, QD, QD, Sc, Sc,
       (long long)Sc * QD, (long long)QD * Sc, (long long)Sc * Sc, BH, stream);
  softmax_kernel<<<BH * Sc, 256, 0, stream>>>(scores, rsqrtf((float)QD));
  // ohead[bh] = scores[bh] (S x S) @ vpack[bh] (S x 128)
  gemm(scores, vpack, ohead, Sc, DVc, Sc, Sc, DVc, DVc,
       (long long)Sc * Sc, (long long)Sc * DVc, (long long)Sc * DVc, BH, stream);
  unpack_o_kernel<<<cdiv(BH * Sc * DVc, 256), 256, 0, stream>>>(ohead, o);
  gemm(o, wo, eo, Tc, Dc, Hc * DVc, Hc * DVc, Dc, Dc, 0, 0, 0, 1, stream); // eo as temp
  add2_kernel<<<cdiv((int)TD, 256), 256, 0, stream>>>(eo, x, first, TD);

  // ---- MoE ----
  rmsnorm_kernel<<<Tc, 256, 0, stream>>>(first, norm_w, h2, Dc, Dc, Dc);
  gemm(h2, gwpad, gates, Tc, GEP, Dc, Dc, GEP, GEP, 0, 0, 0, 1, stream);
  route_kernel<<<cdiv(Tc, 256), 256, 0, stream>>>(gates, gate_b, route);
  fill0_kernel<<<cdiv((int)TD, 256), 256, 0, stream>>>(moeo, TD);

  const long long TF = (long long)Tc * FIc;
  for (int e = 0; e < Ec; ++e) {
    const float* w1e = w1 + (long long)e * Dc * FIc;
    const float* w2e = w2 + (long long)e * FIc * Dc;
    const float* w3e = w3 + (long long)e * Dc * FIc;
    gemm(h2, w1e, hh,  Tc, FIc, Dc, Dc, FIc, FIc, 0, 0, 0, 1, stream);
    gemm(h2, w3e, gg3, Tc, FIc, Dc, Dc, FIc, FIc, 0, 0, 0, 1, stream);
    silu_mul_kernel<<<cdiv((int)TF, 256), 256, 0, stream>>>(hh, gg3, hh, TF);
    gemm(hh, w2e, eo, Tc, Dc, FIc, FIc, Dc, Dc, 0, 0, 0, 1, stream);
    route_accum_kernel<<<cdiv((int)TD, 256), 256, 0, stream>>>(eo, route, moeo, e);
  }

  // shared expert
  const long long TS = (long long)Tc * SIc;
  gemm(h2, ws1, sa, Tc, SIc, Dc, Dc, SIc, SIc, 0, 0, 0, 1, stream);
  gemm(h2, ws3, sb, Tc, SIc, Dc, Dc, SIc, SIc, 0, 0, 0, 1, stream);
  silu_mul_kernel<<<cdiv((int)TS, 256), 256, 0, stream>>>(sa, sb, sa, TS);
  gemm(sa, ws2, sho, Tc, Dc, SIc, SIc, Dc, Dc, 0, 0, 0, 1, stream);

  add3_kernel<<<cdiv((int)TD, 256), 256, 0, stream>>>(moeo, sho, first, out, TD);
}
